// Encoder_Layer_25340307046911
// MI455X (gfx1250) — matrix operand order repaired and checked by reading, not compiled, not run on hardware
//
#include <hip/hip_runtime.h>

#define BS_B 4
#define SEQ  2048
#define DIM  768
#define NH   12
#define DHD  64
#define DFF  3072

typedef __attribute__((ext_vector_type(16))) __bf16 v16bf;
typedef __attribute__((ext_vector_type(8)))  float  v8f;
typedef __attribute__((ext_vector_type(4)))  int    v4i;

union Frag { uint4 u[2]; v16bf v; };

// --- native bf16 converts (v_cvt_*_bf16) ------------------------------------
__device__ __forceinline__ unsigned short f2bf(float f) {
  union { __bf16 h; unsigned short s; } c;
  c.h = (__bf16)f;
  return c.s;
}
__device__ __forceinline__ float bf2f(unsigned short s) {
  return __uint_as_float((unsigned)s << 16);
}
__device__ __forceinline__ unsigned pk2(float lo, float hi) {
  return (unsigned)f2bf(lo) | ((unsigned)f2bf(hi) << 16);
}

// --- async global->LDS (CDNA5), with safe fallback --------------------------
#if __has_builtin(__builtin_amdgcn_global_load_async_to_lds_b128) && \
    __has_builtin(__builtin_amdgcn_s_wait_asynccnt)
#define USE_ASYNC 1
#else
#define USE_ASYNC 0
#endif

__device__ __forceinline__ void async_copy16(const unsigned short* g, unsigned short* l) {
#if USE_ASYNC
  __builtin_amdgcn_global_load_async_to_lds_b128(
      (__attribute__((address_space(1))) v4i*)g,
      (__attribute__((address_space(3))) v4i*)l, 0, 0);
#else
  *(uint4*)l = *(const uint4*)g;
#endif
}
__device__ __forceinline__ void wait_async() {
#if USE_ASYNC
  __builtin_amdgcn_s_wait_asynccnt(0);
#endif
}

// ---------------------------------------------------------------------------
// f32 -> bf16 bulk conversion (one-time pre-pass for weights & inputs)
// ---------------------------------------------------------------------------
__global__ __launch_bounds__(256) void cvt_bf16_kernel(
    const float* __restrict__ src, unsigned short* __restrict__ dst, int n4) {
  int i = blockIdx.x * 256 + threadIdx.x;
  if (i < n4) {
    float4 v = ((const float4*)src)[i];
    uint2 p;
    p.x = pk2(v.x, v.y);
    p.y = pk2(v.z, v.w);
    ((uint2*)dst)[i] = p;
  }
}

// ---------------------------------------------------------------------------
// bf16 WMMA GEMM:  C[M,N] = act(A[M,K] @ W[K,N] + bias)
// Block tile 128x128, 8 waves (4x2), wave tile 32x64 (2x4 WMMA), BK=32.
// Double-buffered LDS; A tile via async DMA, W tile register-transposed.
// OUTF: 0 = bf16 output, 1 = f32 output.  ACT: 1 = exact GELU.
// ---------------------------------------------------------------------------
template <int ACT, int OUTF>
__global__ __launch_bounds__(256) void gemm_bf16_kernel(
    const unsigned short* __restrict__ A, const unsigned short* __restrict__ W,
    const float* __restrict__ bias, void* __restrict__ Cout, int K, int N) {
  __shared__ unsigned short As[2][128 * 32];   // [m][k] bf16
  __shared__ unsigned short Bs[2][128 * 32];   // [n][k] bf16 (transposed)

  const int tid  = threadIdx.x;
  const int lane = tid & 31;
  const int wave = tid >> 5;
  const int wm   = wave >> 1;          // 0..3 -> 32 rows each
  const int wn   = wave & 1;           // 0..1 -> 64 cols each
  const int hi   = lane >> 4;
  const int ln   = lane & 15;
  const size_t blockM = (size_t)blockIdx.x * 128;
  const int    blockN = blockIdx.y * 128;

  auto loadA = [&](int kt, int buf) {
#pragma unroll
    for (int i = 0; i < 2; ++i) {
      int lin = tid + i * 256;               // 0..511
      int row = lin >> 2;
      int c8  = (lin & 3) << 3;
      async_copy16(A + (blockM + row) * K + kt * 32 + c8, &As[buf][row * 32 + c8]);
    }
  };
  auto loadB = [&](int kt, int buf) {
#pragma unroll
    for (int i = 0; i < 2; ++i) {
      int lin = tid + i * 256;               // 0..511
      int kk  = lin >> 4;                    // 0..31
      int n8  = (lin & 15) << 3;             // 0..120
      union { uint4 q; unsigned short s[8]; } w;
      w.q = *(const uint4*)(W + (size_t)(kt * 32 + kk) * N + blockN + n8);
#pragma unroll
      for (int j = 0; j < 8; ++j) Bs[buf][(n8 + j) * 32 + kk] = w.s[j];
    }
  };

  v8f acc[2][4] = {};
  const int ktiles = K >> 5;

  loadA(0, 0);
  loadB(0, 0);
  wait_async();
  __syncthreads();

  for (int kt = 0; kt < ktiles; ++kt) {
    const int buf = kt & 1;
    if (kt + 1 < ktiles) {                   // prefetch next tile into other buffer
      loadA(kt + 1, buf ^ 1);
      loadB(kt + 1, buf ^ 1);
    }

    Frag af[2], bfr[4];
#pragma unroll
    for (int mi = 0; mi < 2; ++mi) {
      const unsigned short* p = &As[buf][(wm * 32 + mi * 16 + ln) * 32 + hi * 8];
      af[mi].u[0] = *(const uint4*)p;        // K = hi*8 .. +7
      af[mi].u[1] = *(const uint4*)(p + 16); // K = 16+hi*8 .. +7
    }
#pragma unroll
    for (int ni = 0; ni < 4; ++ni) {
      const unsigned short* p = &Bs[buf][(wn * 64 + ni * 16 + ln) * 32 + hi * 8];
      bfr[ni].u[0] = *(const uint4*)p;       // K = hi*8 .. +7
      bfr[ni].u[1] = *(const uint4*)(p + 16); // K = 16+hi*8 .. +7
    }
#pragma unroll
    for (int mi = 0; mi < 2; ++mi)
#pragma unroll
      for (int ni = 0; ni < 4; ++ni)
        acc[mi][ni] = __builtin_amdgcn_wmma_f32_16x16x32_bf16(
            false, af[mi].v, false, bfr[ni].v, (short)0, acc[mi][ni], false, false);

    wait_async();
    __syncthreads();
  }

#pragma unroll
  for (int mi = 0; mi < 2; ++mi)
#pragma unroll
    for (int ni = 0; ni < 4; ++ni)
#pragma unroll
      for (int r = 0; r < 8; ++r) {
        size_t row = blockM + wm * 32 + mi * 16 + r + hi * 8;
        int    col = blockN + wn * 64 + ni * 16 + ln;
        float v = acc[mi][ni][r] + bias[col];
        if (ACT == 1) v = 0.5f * v * (1.0f + erff(v * 0.70710678118f));
        if (OUTF)
          ((float*)Cout)[row * N + col] = v;
        else
          ((unsigned short*)Cout)[row * N + col] = f2bf(v);
      }
}

// ---------------------------------------------------------------------------
// Flash attention over bf16 q/k/v; bf16 ctx out.
// Block = (b, h, 128 q rows); wave owns 16 q rows; KV tiles of 32.
// ---------------------------------------------------------------------------
__global__ __launch_bounds__(256) void attn_kernel(
    const unsigned short* __restrict__ qg, const unsigned short* __restrict__ kg,
    const unsigned short* __restrict__ vg, unsigned short* __restrict__ ctx) {
  __shared__ unsigned short Ksm[32 * 64];      // [kv][d] bf16
  __shared__ unsigned short Vsm[64 * 32];      // [d][kv] bf16 (transposed)
  __shared__ unsigned short Psm[8 * 16 * 32];  // per-wave P staging

  const int tid = threadIdx.x, lane = tid & 31, wave = tid >> 5;
  const int hi = lane >> 4, ln = lane & 15;
  const int b = blockIdx.z, h = blockIdx.y;
  const int q0 = blockIdx.x * 128 + wave * 16;

  // q A-fragments straight from global bf16 (no conversion)
  const unsigned short* qp = qg + ((size_t)(b * SEQ + q0 + ln)) * DIM + h * DHD;
  Frag qf[2];
#pragma unroll
  for (int c = 0; c < 2; ++c) {
    int d0 = c * 32 + hi * 8;
    qf[c].u[0] = *(const uint4*)(qp + d0);
    qf[c].u[1] = *(const uint4*)(qp + d0 + 16);
  }

  float m[8], l[8];
#pragma unroll
  for (int r = 0; r < 8; ++r) { m[r] = -1e30f; l[r] = 0.f; }
  v8f accd[4] = {};

  const int kv = tid >> 3;                   // 0..31
  const int c8 = (tid & 7) << 3;             // 0..56

  for (int kt = 0; kt < SEQ / 32; ++kt) {
    const int kvb = kt * 32;
    const unsigned short* krow = kg + ((size_t)(b * SEQ + kvb + kv)) * DIM + h * DHD + c8;
    async_copy16(krow, &Ksm[kv * 64 + c8]);  // K tile: direct row-major DMA

    const unsigned short* vrow = vg + ((size_t)(b * SEQ + kvb + kv)) * DIM + h * DHD + c8;
    union { uint4 q; unsigned short s[8]; } vv;
    vv.q = *(const uint4*)vrow;              // V tile: register transpose
#pragma unroll
    for (int j = 0; j < 8; ++j) Vsm[(c8 + j) * 32 + kv] = vv.s[j];

    wait_async();
    __syncthreads();

    // scores: 16 q rows x 32 kv = two 16x16 WMMA tiles, K over d (2 x 32)
    v8f sc[2];
#pragma unroll
    for (int j = 0; j < 2; ++j) {
      v8f s = {};
#pragma unroll
      for (int c = 0; c < 2; ++c) {
        Frag kf;
        const unsigned short* p = &Ksm[(j * 16 + ln) * 64 + c * 32 + hi * 8];
        kf.u[0] = *(const uint4*)p;
        kf.u[1] = *(const uint4*)(p + 16);
        s = __builtin_amdgcn_wmma_f32_16x16x32_bf16(false, qf[c].v, false, kf.v,
                                                    (short)0, s, false, false);
      }
      sc[j] = s;
    }
    // apply 1/sqrt(DH)
#pragma unroll
    for (int j = 0; j < 2; ++j)
#pragma unroll
      for (int r = 0; r < 8; ++r) sc[j][r] *= 0.125f;

    // online softmax (row stats replicated across each 16-lane half)
    float alpha[8];
#pragma unroll
    for (int r = 0; r < 8; ++r) {
      float v = fmaxf(sc[0][r], sc[1][r]);
#pragma unroll
      for (int o = 1; o < 16; o <<= 1) v = fmaxf(v, __shfl_xor(v, o, 32));
      float mn = fmaxf(m[r], v);
      alpha[r] = __expf(m[r] - mn);
      float p0 = __expf(sc[0][r] - mn);
      float p1 = __expf(sc[1][r] - mn);
      sc[0][r] = p0; sc[1][r] = p1;
      float ps = p0 + p1;
#pragma unroll
      for (int o = 1; o < 16; o <<= 1) ps += __shfl_xor(ps, o, 32);
      l[r] = l[r] * alpha[r] + ps;
      m[r] = mn;
    }
#pragma unroll
    for (int c = 0; c < 4; ++c)
#pragma unroll
      for (int r = 0; r < 8; ++r) accd[c][r] *= alpha[r];

    // P (C layout) -> bf16 -> per-wave LDS -> A-fragment layout
    unsigned short* pp = &Psm[wave * (16 * 32)];
#pragma unroll
    for (int j = 0; j < 2; ++j)
#pragma unroll
      for (int r = 0; r < 8; ++r)
        pp[(r + hi * 8) * 32 + j * 16 + ln] = f2bf(sc[j][r]);
    Frag pf;
    {
      const unsigned short* p = pp + ln * 32 + hi * 8;
      pf.u[0] = *(const uint4*)p;
      pf.u[1] = *(const uint4*)(p + 16);
    }

    // ctx += P @ V  (4 d-chunks of 16)
#pragma unroll
    for (int c = 0; c < 4; ++c) {
      Frag vf;
      const unsigned short* p = &Vsm[(c * 16 + ln) * 32 + hi * 8];
      vf.u[0] = *(const uint4*)p;
      vf.u[1] = *(const uint4*)(p + 16);
      accd[c] = __builtin_amdgcn_wmma_f32_16x16x32_bf16(false, pf.v, false, vf.v,
                                                        (short)0, accd[c], false, false);
    }
    __syncthreads();
  }

  // normalize and write ctx (bf16) into [B,S,D] layout
#pragma unroll
  for (int c = 0; c < 4; ++c)
#pragma unroll
    for (int r = 0; r < 8; ++r) {
      size_t row = (size_t)(b * SEQ) + q0 + r + hi * 8;
      int    col = h * DHD + c * 16 + ln;
      ctx[row * DIM + col] = f2bf(accd[c][r] / l[r]);
    }
}

// ---------------------------------------------------------------------------
// h = LayerNorm(X_f32 + R_bf16) * g + b   -> bf16 out  (one row per block)
// ---------------------------------------------------------------------------
__global__ __launch_bounds__(256) void ln_residual_kernel(
    const float* __restrict__ X, const unsigned short* __restrict__ R,
    const float* __restrict__ g, const float* __restrict__ bta,
    unsigned short* __restrict__ out) {
  __shared__ float red[8];
  __shared__ float stat[2];
  const int tid = threadIdx.x, lane = tid & 31, wave = tid >> 5;
  const size_t row = blockIdx.x;

  float x[3];
  float s = 0.f;
#pragma unroll
  for (int i = 0; i < 3; ++i) {
    int idx = tid + i * 256;
    float v = X[row * DIM + idx] + bf2f(R[row * DIM + idx]);
    x[i] = v; s += v;
  }
#pragma unroll
  for (int o = 1; o < 32; o <<= 1) s += __shfl_xor(s, o, 32);
  if (lane == 0) red[wave] = s;
  __syncthreads();
  if (tid == 0) {
    float t = 0.f;
    for (int i = 0; i < 8; ++i) t += red[i];
    stat[0] = t / (float)DIM;
  }
  __syncthreads();
  const float mu = stat[0];

  float vs = 0.f;
#pragma unroll
  for (int i = 0; i < 3; ++i) { float d = x[i] - mu; vs += d * d; }
#pragma unroll
  for (int o = 1; o < 32; o <<= 1) vs += __shfl_xor(vs, o, 32);
  if (lane == 0) red[wave] = vs;
  __syncthreads();
  if (tid == 0) {
    float t = 0.f;
    for (int i = 0; i < 8; ++i) t += red[i];
    stat[1] = t / (float)DIM;
  }
  __syncthreads();
  const float rstd = rsqrtf(stat[1] + 1e-5f);

#pragma unroll
  for (int i = 0; i < 3; ++i) {
    int idx = tid + i * 256;
    out[row * DIM + idx] = f2bf((x[i] - mu) * rstd * g[idx] + bta[idx]);
  }
}

// ---------------------------------------------------------------------------
extern "C" void kernel_launch(void* const* d_in, const int* in_sizes, int n_in,
                              void* d_out, int out_size, void* d_ws, size_t ws_size,
                              hipStream_t stream) {
  (void)in_sizes; (void)n_in; (void)out_size; (void)ws_size;

  const float* Qin  = (const float*)d_in[0];
  const float* Kin  = (const float*)d_in[1];
  const float* Vin  = (const float*)d_in[2];
  const float* Wq   = (const float*)d_in[3];
  const float* bq   = (const float*)d_in[4];
  const float* Wk   = (const float*)d_in[5];
  const float* bk   = (const float*)d_in[6];
  const float* Wv   = (const float*)d_in[7];
  const float* bv   = (const float*)d_in[8];
  const float* Wo   = (const float*)d_in[9];
  const float* bo   = (const float*)d_in[10];
  const float* ln_g = (const float*)d_in[11];
  const float* ln_b = (const float*)d_in[12];
  const float* W1   = (const float*)d_in[13];
  const float* b1   = (const float*)d_in[14];
  const float* W2   = (const float*)d_in[15];
  const float* b2   = (const float*)d_in[16];
  float* out = (float*)d_out;

  const size_t M  = (size_t)BS_B * SEQ;     // 8192
  const size_t SZ = M * DIM;                // 6291456
  const size_t WW = (size_t)DIM * DIM;      // 589824
  const size_t WF = (size_t)DIM * DFF;      // 2359296

  unsigned short* p = (unsigned short*)d_ws;
  unsigned short* Qb  = p; p += SZ;
  unsigned short* Kb  = p; p += SZ;
  unsigned short* Vb  = p; p += SZ;
  unsigned short* WqB = p; p += WW;
  unsigned short* WkB = p; p += WW;
  unsigned short* WvB = p; p += WW;
  unsigned short* WoB = p; p += WW;
  unsigned short* W1B = p; p += WF;
  unsigned short* W2B = p; p += WF;
  unsigned short* qb  = p; p += SZ;
  unsigned short* kb  = p; p += SZ;
  unsigned short* vb  = p; p += SZ;
  unsigned short* ctx = p; p += SZ;
  unsigned short* aob = p; p += SZ;
  unsigned short* hb  = p; p += SZ;
  unsigned short* ff1 = p;                  // M * DFF

  dim3 blk(256);
  auto cvt = [&](const float* s, unsigned short* d, size_t n) {
    int n4 = (int)(n >> 2);
    cvt_bf16_kernel<<<dim3((n4 + 255) / 256), blk, 0, stream>>>(s, d, n4);
  };
  // one-time bf16 conversion of inputs + weights
  cvt(Qin, Qb, SZ); cvt(Kin, Kb, SZ); cvt(Vin, Vb, SZ);
  cvt(Wq, WqB, WW); cvt(Wk, WkB, WW); cvt(Wv, WvB, WW); cvt(Wo, WoB, WW);
  cvt(W1, W1B, WF); cvt(W2, W2B, WF);

  dim3 gD((unsigned)(M / 128), DIM / 128);  // 64 x 6
  gemm_bf16_kernel<0, 0><<<gD, blk, 0, stream>>>(Qb, WqB, bq, qb, DIM, DIM);
  gemm_bf16_kernel<0, 0><<<gD, blk, 0, stream>>>(Kb, WkB, bk, kb, DIM, DIM);
  gemm_bf16_kernel<0, 0><<<gD, blk, 0, stream>>>(Vb, WvB, bv, vb, DIM, DIM);

  attn_kernel<<<dim3(SEQ / 128, NH, BS_B), blk, 0, stream>>>(qb, kb, vb, ctx);

  gemm_bf16_kernel<0, 0><<<gD, blk, 0, stream>>>(ctx, WoB, bo, aob, DIM, DIM);
  ln_residual_kernel<<<dim3((unsigned)M), blk, 0, stream>>>(Qin, aob, ln_g, ln_b, hb);

  gemm_bf16_kernel<1, 0><<<dim3((unsigned)(M / 128), DFF / 128), blk, 0, stream>>>(
      hb, W1B, b1, ff1, DIM, DFF);
  gemm_bf16_kernel<0, 1><<<gD, blk, 0, stream>>>(ff1, W2B, b2, out, DFF, DIM);
}
